// MultiHeadedSelfAttention_73340861546952
// MI455X (gfx1250) — matrix operand order repaired and checked by reading, not compiled, not run on hardware
//
#include <hip/hip_runtime.h>
#include <hip/hip_bf16.h>

// CDNA5 / gfx1250, wave32. WMMA bf16 path with f32 accumulation.
typedef __attribute__((ext_vector_type(16))) __bf16 v16bf;
typedef __attribute__((ext_vector_type(8)))  __bf16 v8bf;
typedef __attribute__((ext_vector_type(4)))  __bf16 v4bf;
typedef __attribute__((ext_vector_type(8)))  float  v8f;

#define Bsz 2
#define Ssz 2048
#define Dsz 1024
#define Hn  16
#define HDs 64

// ---------------------------------------------------------------------------
// Pack x (fp32 [B*S, D]) -> bf16 xb [B*S, D]. 4 elems/thread, vector I/O.
// ---------------------------------------------------------------------------
__global__ __launch_bounds__(256) void pack_x_kernel(
    const float* __restrict__ x, __bf16* __restrict__ xb)
{
  const size_t i = ((size_t)blockIdx.x * blockDim.x + threadIdx.x) * 4;
  const float4 u = *reinterpret_cast<const float4*>(x + i);
  v4bf o;
  o[0] = (__bf16)u.x; o[1] = (__bf16)u.y; o[2] = (__bf16)u.z; o[3] = (__bf16)u.w;
  *reinterpret_cast<v4bf*>(xb + i) = o;
}

// ---------------------------------------------------------------------------
// Pack+transpose weights: w[k][n] fp32 -> wt[mat][n][k] bf16.
// Coalesced fp32 reads (n contiguous); 2B strided writes land in L2.
// ---------------------------------------------------------------------------
__global__ __launch_bounds__(256) void pack_w_kernel(
    const float* __restrict__ wq, const float* __restrict__ wk,
    const float* __restrict__ wv, __bf16* __restrict__ wt)
{
  const int t   = blockIdx.x * blockDim.x + threadIdx.x;   // 0 .. 3M-1
  const int mat = t >> 20;
  const int r   = t & 0xFFFFF;                              // k*1024 + n
  const int k   = r >> 10;
  const int n   = r & 1023;
  const float* w = (mat == 0) ? wq : (mat == 1) ? wk : wv;
  wt[((size_t)mat << 20) + ((size_t)n << 10) + k] = (__bf16)w[r];
}

// ---------------------------------------------------------------------------
// Kernel 1: fused QKV projection + bias + RoPE from pre-packed bf16 operands.
// One wave computes a 16(M) x 64(N) tile (the supertile == one head of q/k/v).
// Double-buffered fragments: chunk kc+1 is in flight while chunk kc's four
// WMMAs execute, so the compiler can issue partial loadcnt waits instead of
// s_wait_loadcnt 0 before every WMMA.
// q,k stored [B,H,S,HD]; v stored TRANSPOSED [B,H,HD,S] so the attention
// kernel's P.V B-fragment is a contiguous 32B load.
// ---------------------------------------------------------------------------
__global__ __launch_bounds__(256) void qkv_rope_kernel(
    const __bf16* __restrict__ xb, const __bf16* __restrict__ wt,
    const float* __restrict__ bq, const float* __restrict__ bk,
    const float* __restrict__ bv,
    __bf16* __restrict__ q_ws, __bf16* __restrict__ k_ws,
    __bf16* __restrict__ vt_ws)
{
  const int lane = threadIdx.x & 31;
  const int ln   = lane & 15;
  const int half = lane >> 4;
  const int wid  = blockIdx.x * (blockDim.x >> 5) + (threadIdx.x >> 5);

  const int mtile  = wid / 48;         // 0..255  (B*S/16 row tiles)
  const int nsuper = wid % 48;         // 0..47   (3 matrices * 16 heads)
  const int mat    = nsuper >> 4;      // 0=q 1=k 2=v
  const int h      = nsuper & 15;

  const __bf16* w    = wt + ((size_t)mat << 20);
  const float*  bias = (mat == 0) ? bq : (mat == 1) ? bk : bv;

  const int arow = mtile * 16 + ln;    // A-matrix: lane holds row m = ln
  const __bf16* apbase = xb + (size_t)arow * Dsz + 16 * half;
  const __bf16* bp[4];
  #pragma unroll
  for (int nt = 0; nt < 4; ++nt)
    bp[nt] = w + ((size_t)(h * HDs + nt * 16 + ln) << 10) + 16 * half;

  v8f acc[4];
  #pragma unroll
  for (int nt = 0; nt < 4; ++nt)
    #pragma unroll
    for (int r = 0; r < 8; ++r) acc[nt][r] = 0.0f;

  // A fragment 16x32 bf16: k = 16*half+i
  auto loadA = [&](int kc, v16bf& f) {
    const __bf16* ap = apbase + kc * 32;
    const v8bf a0 = *reinterpret_cast<const v8bf*>(ap);
    const v8bf a1 = *reinterpret_cast<const v8bf*>(ap + 8);
    #pragma unroll
    for (int i = 0; i < 8; ++i) { f[i] = a0[i]; f[8 + i] = a1[i]; }
  };

  v16bf aF[2];
  v16bf bF[2][4];
  loadA(0, aF[0]);
  #pragma unroll
  for (int nt = 0; nt < 4; ++nt)
    bF[0][nt] = *reinterpret_cast<const v16bf*>(bp[nt]);

  for (int kc = 0; kc < Dsz / 32; ++kc) {
    const int cur = kc & 1, nxt = cur ^ 1;
    if (kc + 1 < Dsz / 32) {                       // prefetch next chunk
      loadA(kc + 1, aF[nxt]);
      #pragma unroll
      for (int nt = 0; nt < 4; ++nt)
        bF[nxt][nt] = *reinterpret_cast<const v16bf*>(bp[nt] + (kc + 1) * 32);
    }
    #pragma unroll
    for (int nt = 0; nt < 4; ++nt)
      acc[nt] = __builtin_amdgcn_wmma_f32_16x16x32_bf16(
          false, aF[cur], false, bF[cur][nt], (short)0, acc[nt], false, false);
  }

  // Epilogue
  const int grow0 = mtile * 16 + 8 * half;   // first C row this lane holds
  const int bidx  = grow0 >> 11;             // / S (tiles never straddle batch)
  const int s0    = grow0 & (Ssz - 1);

  if (mat == 2) {
    // V: bias only, store transposed [B,H,HD,S]; rows are consecutive s ->
    // contiguous v8bf (16B) store per n-tile.
    #pragma unroll
    for (int nt = 0; nt < 4; ++nt) {
      const int hd = nt * 16 + ln;
      const float bv_ = bias[h * HDs + hd];
      v8bf pack;
      #pragma unroll
      for (int r = 0; r < 8; ++r) pack[r] = (__bf16)(acc[nt][r] + bv_);
      *reinterpret_cast<v8bf*>(
          vt_ws + ((size_t)((bidx * Hn + h) * HDs + hd)) * Ssz + s0) = pack;
    }
  } else {
    __bf16* dst = (mat == 0) ? q_ws : k_ws;
    #pragma unroll
    for (int nt = 0; nt < 4; ++nt) {
      const int hd = nt * 16 + ln;
      const float bv_ = bias[h * HDs + hd];
      // theta = 10000^(-(hd&~1)/64)
      const float theta = __expf(-(float)(hd & ~1) * (9.210340372f / 64.0f));
      #pragma unroll
      for (int r = 0; r < 8; ++r) {
        const int s = s0 + r;
        float val = acc[nt][r] + bv_;
        float sn, cs;
        __sincosf((float)s * theta, &sn, &cs);
        const float pv = __shfl_xor(val, 1);        // pair partner column hd^1
        val = (hd & 1) ? (val * cs + pv * sn)       // o1 = t1*c + t0*s
                       : (val * cs - pv * sn);      // o0 = t0*c - t1*s
        dst[(((size_t)(bidx * Hn + h)) * Ssz + s) * HDs + hd] = (__bf16)val;
      }
    }
  }
}

// ---------------------------------------------------------------------------
// Kernel 2: banded (causal + history + key-padding) flash attention.
// One wave per 16-query tile of one (b,h). Keys in chunks of 32:
//   QK^T: 2 n-tiles x 2 k-chunks = 4 wmma ; P.V: 4 hd-tiles = 4 wmma.
// All independent loads (K, V, padding) are issued at the top of the chunk so
// V fragments stay in flight across the whole softmax VALU sequence. P goes
// C-layout -> A-layout through per-wave LDS with an explicit s_wait_dscnt
// (no block barrier, since waves have different band trip counts).
// ---------------------------------------------------------------------------
__global__ __launch_bounds__(128) void attn_kernel(
    const __bf16* __restrict__ q_ws, const __bf16* __restrict__ k_ws,
    const __bf16* __restrict__ vt_ws,
    const int* __restrict__ kpm, const int* __restrict__ mhl,
    float* __restrict__ out)
{
  __shared__ __align__(32) __bf16 Plds[4][16][32];  // per-wave 16x32 P tile

  const int lane  = threadIdx.x & 31;
  const int ln    = lane & 15;
  const int half  = lane >> 4;
  const int wslot = threadIdx.x >> 5;
  const int wid   = blockIdx.x * 4 + wslot;

  const int qtile = wid & 127;         // S/16 = 128
  const int h     = (wid >> 7) & 15;
  const int b     = wid >> 11;
  const int q0    = qtile * 16;
  const int hist  = mhl[0];

  const __bf16* qp  = q_ws  + (size_t)(b * Hn + h) * Ssz * HDs;
  const __bf16* kp  = k_ws  + (size_t)(b * Hn + h) * Ssz * HDs;
  const __bf16* vtp = vt_ws + (size_t)(b * Hn + h) * HDs * Ssz;

  // Q as two A fragments over HD=64
  v16bf aq[2];
  #pragma unroll
  for (int kc = 0; kc < 2; ++kc) {
    const __bf16* p = qp + (size_t)(q0 + ln) * HDs + kc * 32 + 16 * half;
    const v8bf a0 = *reinterpret_cast<const v8bf*>(p);
    const v8bf a1 = *reinterpret_cast<const v8bf*>(p + 8);
    #pragma unroll
    for (int i = 0; i < 8; ++i) { aq[kc][i] = a0[i]; aq[kc][8 + i] = a1[i]; }
  }

  v8f  o[4];
  float M[8], L[8];
  #pragma unroll
  for (int nt = 0; nt < 4; ++nt)
    #pragma unroll
    for (int r = 0; r < 8; ++r) o[nt][r] = 0.0f;
  #pragma unroll
  for (int r = 0; r < 8; ++r) { M[r] = -1e30f; L[r] = 0.0f; }

  int jmin = q0 - (hist - 1); if (jmin < 0) jmin = 0;
  const int kb0 = jmin & ~31;

  for (int kb = kb0; kb <= q0 + 15; kb += 32) {
    // ---- issue ALL independent loads first: K frags, V frags, padding
    v16bf kf[2][2];                                   // [n-tile][k-chunk]
    #pragma unroll
    for (int nt = 0; nt < 2; ++nt) {
      const int j = kb + nt * 16 + ln;                // B-frag column = key
      #pragma unroll
      for (int kc = 0; kc < 2; ++kc)
        kf[nt][kc] = *reinterpret_cast<const v16bf*>(
            kp + (size_t)j * HDs + kc * 32 + 16 * half);
    }
    v16bf vb[4];                                      // V^T -> contiguous 32B
    #pragma unroll
    for (int nt = 0; nt < 4; ++nt)
      vb[nt] = *reinterpret_cast<const v16bf*>(
          vtp + (size_t)(nt * 16 + ln) * Ssz + kb + 16 * half);
    const int pad0 = kpm[b * Ssz + (kb + ln)];
    const int pad1 = kpm[b * Ssz + (kb + 16 + ln)];

    // ---- scores: Q (16x64) . K^T (64x32) -> two 16x16 f32 fragments
    v8f sc[2];
    #pragma unroll
    for (int nt = 0; nt < 2; ++nt) {
      #pragma unroll
      for (int r = 0; r < 8; ++r) sc[nt][r] = 0.0f;
      #pragma unroll
      for (int kc = 0; kc < 2; ++kc)
        sc[nt] = __builtin_amdgcn_wmma_f32_16x16x32_bf16(
            false, aq[kc], false, kf[nt][kc], (short)0, sc[nt], false, false);
    }

    // ---- scale + band/padding mask (per-lane selects; uniform control flow)
    #pragma unroll
    for (int nt = 0; nt < 2; ++nt) {
      const int j  = kb + nt * 16 + ln;
      const int pd = nt ? pad1 : pad0;
      #pragma unroll
      for (int r = 0; r < 8; ++r) {
        const int i = q0 + r + 8 * half;
        const bool ok = (j <= i) && ((i - j) < hist) && (pd == 0);
        sc[nt][r] = ok ? sc[nt][r] * 0.125f : -1e30f;  // 1/sqrt(64)
      }
    }

    // ---- online softmax (row reductions within each 16-lane half)
    float sf[8];
    #pragma unroll
    for (int r = 0; r < 8; ++r) {
      float cm = fmaxf(sc[0][r], sc[1][r]);
      #pragma unroll
      for (int m = 1; m < 16; m <<= 1) cm = fmaxf(cm, __shfl_xor(cm, m));
      const float Mn = fmaxf(M[r], cm);
      sf[r] = __expf(M[r] - Mn);
      M[r]  = Mn;
    }

    #pragma unroll
    for (int r = 0; r < 8; ++r) {
      const float p0 = (sc[0][r] < -1e29f) ? 0.0f : __expf(sc[0][r] - M[r]);
      const float p1 = (sc[1][r] < -1e29f) ? 0.0f : __expf(sc[1][r] - M[r]);
      Plds[wslot][r + 8 * half][ln]      = (__bf16)p0;
      Plds[wslot][r + 8 * half][16 + ln] = (__bf16)p1;
      float rs = p0 + p1;
      #pragma unroll
      for (int m = 1; m < 16; m <<= 1) rs += __shfl_xor(rs, m);
      L[r] = L[r] * sf[r] + rs;
      #pragma unroll
      for (int nt = 0; nt < 4; ++nt) o[nt][r] *= sf[r];
    }

    // per-wave LDS RAW: drain DS pipe (CDNA5 split counter)
    asm volatile("s_wait_dscnt 0x0" ::: "memory");

    // ---- re-read P as A fragment (16x32, K = keys)
    v16bf pa;
    {
      const __bf16* pr = &Plds[wslot][ln][0];
      const v8bf p0 = *reinterpret_cast<const v8bf*>(pr + 16 * half);
      const v8bf p1 = *reinterpret_cast<const v8bf*>(pr + 8 + 16 * half);
      #pragma unroll
      for (int i = 0; i < 8; ++i) { pa[i] = p0[i]; pa[8 + i] = p1[i]; }
    }

    // ---- O += P (16x32) . V (32x64)
    #pragma unroll
    for (int nt = 0; nt < 4; ++nt)
      o[nt] = __builtin_amdgcn_wmma_f32_16x16x32_bf16(
          false, pa, false, vb[nt], (short)0, o[nt], false, false);
  }

  // ---- normalize + store out[b, s, h*64+hd] (fp32)
  #pragma unroll
  for (int nt = 0; nt < 4; ++nt) {
    const int hd = nt * 16 + ln;
    #pragma unroll
    for (int r = 0; r < 8; ++r) {
      const int s = q0 + r + 8 * half;
      out[((size_t)(b * Ssz + s)) * (Hn * HDs) + h * HDs + hd] = o[nt][r] / L[r];
    }
  }
}

// ---------------------------------------------------------------------------
extern "C" void kernel_launch(void* const* d_in, const int* in_sizes, int n_in,
                              void* d_out, int out_size, void* d_ws, size_t ws_size,
                              hipStream_t stream) {
  const float* x   = (const float*)d_in[0];
  const float* wq  = (const float*)d_in[1];
  const float* bq  = (const float*)d_in[2];
  const float* wk  = (const float*)d_in[3];
  const float* bk  = (const float*)d_in[4];
  const float* wv  = (const float*)d_in[5];
  const float* bv  = (const float*)d_in[6];
  const int*   kpm = (const int*)d_in[7];
  const int*   mhl = (const int*)d_in[8];
  float*       out = (float*)d_out;

  const size_t qkv_elems = (size_t)Bsz * Hn * Ssz * HDs;   // 4M bf16 each
  const size_t x_elems   = (size_t)Bsz * Ssz * Dsz;        // 4M bf16
  const size_t w_elems   = (size_t)3 * Dsz * Hn * HDs;     // 3M bf16

  __bf16* q_ws  = (__bf16*)d_ws;
  __bf16* k_ws  = q_ws  + qkv_elems;
  __bf16* vt_ws = k_ws  + qkv_elems;
  __bf16* xb    = vt_ws + qkv_elems;
  __bf16* wt    = xb    + x_elems;
  // total: (3*4M + 4M + 3M) * 2B = 38 MB of d_ws

  // One-time bf16 repack (bandwidth-trivial; everything then L2-resident)
  pack_x_kernel<<<(int)(x_elems / 4 / 256), 256, 0, stream>>>(x, xb);
  pack_w_kernel<<<(int)(w_elems / 256), 256, 0, stream>>>(wq, wk, wv, wt);

  // 256 M-tiles * 48 N-supertiles = 12288 waves, 8 waves/block
  qkv_rope_kernel<<<1536, 256, 0, stream>>>(xb, wt, bq, bk, bv,
                                            q_ws, k_ws, vt_ws);
  // B*H*(S/16) = 4096 waves, 4 waves/block
  attn_kernel<<<1024, 128, 0, stream>>>(q_ws, k_ws, vt_ws, kpm, mhl, out);
}
